// DecoupledDynamics_549755813933
// MI455X (gfx1250) — matrix operand order repaired and checked by reading, not compiled, not run on hardware
//
#include <hip/hip_runtime.h>
#include <stdint.h>

// ---------------------------------------------------------------------------
// DecoupledDynamics fused MoE-MLP for gfx1250 (MI455X, wave32, WMMA).
//
// Strategy: bucket samples by policy index, then run a fused
//   y = relu(x @ W1[p] + b1[p]) @ W2[p] + b2[p]
// grouped-GEMM using V_WMMA_F32_16X16X32_BF16 with fp32 emulated via
// (hi,lo) bf16 split (3 WMMAs per tile product -> ~fp32 accuracy).
//
// Workspace layout (requires ws_size >= ~36 MB):
//   [0,32)        counts[8]
//   [64,96)       cursor[8]
//   [128,164)     abase[9]        (64-aligned bucket starts + total)
//   [4096,..)     perm[16384+512] (bucketed sample ids, -1 = pad)
//   [OFF_W1C,..)  W1 as hi/lo bf16, WMMA-B-operand swizzled  (18.9 MB)
//   [OFF_W2C,..)  W2 as hi/lo bf16, WMMA-B-operand swizzled  (16.8 MB)
// ---------------------------------------------------------------------------

#define BN   16384
#define PP   8
#define DD   512
#define AAd  64
#define KXd  576          // D + A
#define HHd  1024

#define OFF_COUNTS 0
#define OFF_CURSOR 64
#define OFF_ABASE  128
#define OFF_PERM   4096
#define PERM_N     (BN + PP*64)
#define OFF_W1C    (131072)                 // 8*576*1024*4 = 18874368 bytes
#define OFF_W2C    (19005440)               // 8*1024*512*4 = 16777216 bytes

typedef __attribute__((ext_vector_type(16))) __bf16 v16bf;
typedef __attribute__((ext_vector_type(8)))  float  v8f;

#define WMMA_BF16(a, b, c) \
  __builtin_amdgcn_wmma_f32_16x16x32_bf16(false, (a), false, (b), (short)0, (c), false, false)

__device__ __forceinline__ unsigned short f32_to_bf16_bits(float f) {
  unsigned u = __builtin_bit_cast(unsigned, f);
  u += 0x7FFFu + ((u >> 16) & 1u);          // round-to-nearest-even
  return (unsigned short)(u >> 16);
}
__device__ __forceinline__ float bf16_bits_to_f32(unsigned short s) {
  return __builtin_bit_cast(float, ((unsigned)s) << 16);
}

// ---------------------------------------------------------------------------
// Bucketing kernels
// ---------------------------------------------------------------------------
__global__ void init_kernel(int* counts, int* perm) {
  int i = blockIdx.x * blockDim.x + threadIdx.x;
  if (i < PP) counts[i] = 0;
  if (i < PERM_N) perm[i] = -1;
}

__global__ void histo_kernel(const int* __restrict__ pidx, int* counts) {
  __shared__ int lc[PP];
  if (threadIdx.x < PP) lc[threadIdx.x] = 0;
  __syncthreads();
  int b = blockIdx.x * blockDim.x + threadIdx.x;
  if (b < BN) atomicAdd(&lc[pidx[b] & 7], 1);
  __syncthreads();
  if (threadIdx.x < PP) atomicAdd(&counts[threadIdx.x], lc[threadIdx.x]);
}

__global__ void prefix_kernel(const int* counts, int* cursor, int* abase) {
  if (blockIdx.x == 0 && threadIdx.x == 0) {
    int acc = 0;
    abase[0] = 0;
    for (int q = 0; q < PP; ++q) {
      cursor[q] = acc;
      acc += ((counts[q] + 63) >> 6) << 6;   // 64-align each bucket
      abase[q + 1] = acc;
    }
  }
}

__global__ void scatter_kernel(const int* __restrict__ pidx, int* cursor,
                               int* __restrict__ perm) {
  int b = blockIdx.x * blockDim.x + threadIdx.x;
  if (b < BN) {
    int p = pidx[b] & 7;
    int pos = atomicAdd(&cursor[p], 1);
    perm[pos] = b;
  }
}

// ---------------------------------------------------------------------------
// Weight conversion: fp32 -> (hi,lo) bf16, pre-swizzled into the WMMA
// B-operand (32xK x 16xN bf16) lane layout. Per 32x16 tile: 2048 bytes =
// hi-plane [32 lanes x 8 dwords] then lo-plane. Lane l holds 32 contiguous
// bytes (its v16bf operand). B layout: lanes 0-15 hold K=0..7,16..23, lanes 16-31
// hold K=8..15,24..31, two K per dword (low half = even K).
// ---------------------------------------------------------------------------
__global__ void convert_weights_kernel(const float* __restrict__ src,
                                       uint32_t* __restrict__ dst,
                                       int KT, int NT, int Kdim, int Ndim) {
  int t = blockIdx.x * blockDim.x + threadIdx.x;
  int total = PP * KT * NT * 256;
  if (t >= total) return;
  int idx = t & 255, tile = t >> 8;
  int kt = tile % KT;
  int rem = tile / KT;
  int nt = rem % NT;
  int p  = rem / NT;                          // tile = (p*NT + nt)*KT + kt
  int lane = idx >> 3, v = idx & 7;
  int kgrp = lane >> 4, nn = lane & 15;
  int kk = ((v & 4) << 2) + ((v & 3) << 1) + (kgrp << 3);   // B-operand layout
  int K0 = kt * 32 + kk, N0 = nt * 16 + nn;
  const float* base = src + (size_t)p * Kdim * Ndim;
  float w0 = base[(size_t)K0 * Ndim + N0];
  float w1 = base[(size_t)(K0 + 1) * Ndim + N0];
  unsigned short h0 = f32_to_bf16_bits(w0), h1 = f32_to_bf16_bits(w1);
  unsigned short l0 = f32_to_bf16_bits(w0 - bf16_bits_to_f32(h0));
  unsigned short l1 = f32_to_bf16_bits(w1 - bf16_bits_to_f32(h1));
  uint32_t tb = (uint32_t)tile * 512;         // dwords per 2048B tile block
  dst[tb + lane * 8 + v]       = ((uint32_t)h1 << 16) | h0;   // hi plane
  dst[tb + 256 + lane * 8 + v] = ((uint32_t)l1 << 16) | l0;   // lo plane
}

// ---------------------------------------------------------------------------
// Fused MLP kernel. 64 rows per block, 16 waves (512 threads, wave32).
// LDS: [0,256) rowSample[64]; [256, 256+147456) x tile in A layout (hi/lo);
//      then 64KB hidden-chunk in A layout (hi/lo).  Total 213248 B (<320KB).
// ---------------------------------------------------------------------------
#define XTK   18                       // 576/32 K-tiles of x
#define XOFF  256
#define XBYTES (4 * XTK * 2048)        // 147456
#define HOFF  (XOFF + XBYTES)          // 147712
#define LDSB  (HOFF + 4 * 8 * 2048)    // 213248

__global__ __launch_bounds__(512)
void mlp_kernel(const float* __restrict__ latents, const float* __restrict__ actions,
                const float* __restrict__ b1, const float* __restrict__ b2,
                const char* __restrict__ w1c, const char* __restrict__ w2c,
                const int* __restrict__ abase, const int* __restrict__ perm,
                float* __restrict__ out) {
  extern __shared__ __attribute__((aligned(1024))) char sm[];
  int* rowSample = (int*)sm;

  const int rowStart = blockIdx.x * 64;
  if (rowStart >= abase[8]) return;
  int p = 0;
#pragma unroll
  for (int q = 1; q < PP; ++q)
    if (rowStart >= abase[q]) p = q;

  const int tid = threadIdx.x;
  if (tid < 64) rowSample[tid] = perm[rowStart + tid];
  __syncthreads();

  // ---- gather x rows, split to hi/lo bf16, swizzle into A-operand layout --
  // A layout (16-bit, 16x32): lane = mm + 16*kgrp; dword v: K = (v>=4?16:0)
  // + 2*(v&3) + 8*kgrp, pair (K, K+1).
  for (int s = tid; s < 4 * XTK * 256; s += 512) {
    int v = s & 7, lane = (s >> 3) & 31, tile = s >> 8;   // tile = mt*XTK+kt
    int kt = tile % XTK, mt = tile / XTK;
    int kgrp = lane >> 4, mm = lane & 15;
    int kk = ((v & 4) << 2) + ((v & 3) << 1) + (kgrp << 3);
    int m = mt * 16 + mm, k0 = kt * 32 + kk;
    int smp = rowSample[m];
    float w0 = 0.f, w1 = 0.f;
    if (smp >= 0) {
      w0 = (k0 < DD) ? latents[(size_t)smp * DD + k0]
                     : actions[(size_t)smp * AAd + (k0 - DD)];
      w1 = (k0 + 1 < DD) ? latents[(size_t)smp * DD + k0 + 1]
                         : actions[(size_t)smp * AAd + (k0 + 1 - DD)];
    }
    unsigned short h0 = f32_to_bf16_bits(w0), h1 = f32_to_bf16_bits(w1);
    unsigned short l0 = f32_to_bf16_bits(w0 - bf16_bits_to_f32(h0));
    unsigned short l1 = f32_to_bf16_bits(w1 - bf16_bits_to_f32(h1));
    uint32_t* dst = (uint32_t*)(sm + XOFF + tile * 2048);
    dst[lane * 8 + v]       = ((uint32_t)h1 << 16) | h0;
    dst[256 + lane * 8 + v] = ((uint32_t)l1 << 16) | l0;
  }
  __syncthreads();

  const int w = tid >> 5;          // wave id 0..15
  const int lane = tid & 31;
  const int nn = lane & 15;
  const int mmHalf = (lane >> 4) << 3;   // C/D: M = r + (lane>=16 ? 8 : 0)

  v8f yacc[8] = {};                // 64x512 f32 output accumulators (block)

  for (int hc = 0; hc < 4; ++hc) {               // H chunks of 256 columns
    // ---------------- phase 1: h_chunk = relu(x @ W1 + b1) ----------------
    const int nt1g = hc * 16 + w;                // this wave's H n-tile
    v8f hacc[4] = {};
    for (int kt = 0; kt < XTK; ++kt) {
      const char* wb =
          w1c + (size_t)((p * 64 + nt1g) * XTK + kt) * 2048 + lane * 32;
      v16bf bhi = *(const v16bf*)(wb);
      v16bf blo = *(const v16bf*)(wb + 1024);
#pragma unroll
      for (int mt = 0; mt < 4; ++mt) {
        const char* ab = sm + XOFF + (mt * XTK + kt) * 2048 + lane * 32;
        v16bf ahi = *(const v16bf*)(ab);
        v16bf alo = *(const v16bf*)(ab + 1024);
        hacc[mt] = WMMA_BF16(ahi, bhi, hacc[mt]);
        hacc[mt] = WMMA_BF16(ahi, blo, hacc[mt]);
        hacc[mt] = WMMA_BF16(alo, bhi, hacc[mt]);
      }
    }
    const float b1v = b1[p * HHd + nt1g * 16 + nn];
#pragma unroll
    for (int mt = 0; mt < 4; ++mt) {
#pragma unroll
      for (int r = 0; r < 8; ++r) {
        float val = hacc[mt][r] + b1v;
        val = val > 0.f ? val : 0.f;
        unsigned short hb16 = f32_to_bf16_bits(val);
        unsigned short lb16 = f32_to_bf16_bits(val - bf16_bits_to_f32(hb16));
        int mm = r + mmHalf;
        int kk2 = (w << 4) + nn;                 // column within 256-chunk
        int kt2 = kk2 >> 5, kkl = kk2 & 31;
        int lane2 = mm + ((kkl & 8) ? 16 : 0);
        int v2 = ((kkl & 16) >> 2) + ((kkl & 7) >> 1);
        char* hb = sm + HOFF + (mt * 8 + kt2) * 2048 + lane2 * 32 + v2 * 4 +
                   (kkl & 1) * 2;
        *(unsigned short*)hb = hb16;
        *(unsigned short*)(hb + 1024) = lb16;
      }
    }
    __syncthreads();

    // ---------------- phase 2: y += h_chunk @ W2[chunk rows] --------------
    for (int kt2 = 0; kt2 < 8; ++kt2) {
      const int ktg = hc * 8 + kt2;
      v16bf ahi[4], alo[4];
#pragma unroll
      for (int mt = 0; mt < 4; ++mt) {
        const char* ab = sm + HOFF + (mt * 8 + kt2) * 2048 + lane * 32;
        ahi[mt] = *(const v16bf*)(ab);
        alo[mt] = *(const v16bf*)(ab + 1024);
      }
#pragma unroll
      for (int ni = 0; ni < 2; ++ni) {
        const int nt2 = 2 * w + ni;              // this wave's D n-tiles
        const char* wb =
            w2c + (size_t)((p * 32 + nt2) * 32 + ktg) * 2048 + lane * 32;
        v16bf bhi = *(const v16bf*)(wb);
        v16bf blo = *(const v16bf*)(wb + 1024);
#pragma unroll
        for (int mt = 0; mt < 4; ++mt) {
          yacc[mt * 2 + ni] = WMMA_BF16(ahi[mt], bhi, yacc[mt * 2 + ni]);
          yacc[mt * 2 + ni] = WMMA_BF16(ahi[mt], blo, yacc[mt * 2 + ni]);
          yacc[mt * 2 + ni] = WMMA_BF16(alo[mt], bhi, yacc[mt * 2 + ni]);
        }
      }
    }
    __syncthreads();   // protect h-chunk LDS before next phase-1 overwrite
  }

  // ---------------- epilogue: + b2, scatter rows back -----------------------
#pragma unroll
  for (int ni = 0; ni < 2; ++ni) {
    const int nt2 = 2 * w + ni;
    const float b2v = b2[p * DD + nt2 * 16 + nn];
#pragma unroll
    for (int mt = 0; mt < 4; ++mt) {
#pragma unroll
      for (int r = 0; r < 8; ++r) {
        int smp = rowSample[mt * 16 + r + mmHalf];
        if (smp >= 0)
          out[(size_t)smp * DD + nt2 * 16 + nn] = yacc[mt * 2 + ni][r] + b2v;
      }
    }
  }
}

// ---------------------------------------------------------------------------
extern "C" void kernel_launch(void* const* d_in, const int* in_sizes, int n_in,
                              void* d_out, int out_size, void* d_ws, size_t ws_size,
                              hipStream_t stream) {
  const float* latents = (const float*)d_in[0];
  const float* actions = (const float*)d_in[1];
  const int*   pidx    = (const int*)d_in[2];
  const float* W1      = (const float*)d_in[3];
  const float* b1      = (const float*)d_in[4];
  const float* W2      = (const float*)d_in[5];
  const float* b2      = (const float*)d_in[6];
  float* out = (float*)d_out;
  char* ws = (char*)d_ws;

  int* counts = (int*)(ws + OFF_COUNTS);
  int* cursor = (int*)(ws + OFF_CURSOR);
  int* abase  = (int*)(ws + OFF_ABASE);
  int* perm   = (int*)(ws + OFF_PERM);
  uint32_t* w1c = (uint32_t*)(ws + OFF_W1C);
  uint32_t* w2c = (uint32_t*)(ws + OFF_W2C);

  // 1) bucket samples by policy (64-aligned buckets, -1 padding)
  init_kernel<<<(PERM_N + 255) / 256, 256, 0, stream>>>(counts, perm);
  histo_kernel<<<BN / 256, 256, 0, stream>>>(pidx, counts);
  prefix_kernel<<<1, 64, 0, stream>>>(counts, cursor, abase);
  scatter_kernel<<<BN / 256, 256, 0, stream>>>(pidx, cursor, perm);

  // 2) split + swizzle weights into WMMA B-operand hi/lo bf16 planes
  convert_weights_kernel<<<(PP * 18 * 64 * 256) / 256, 256, 0, stream>>>(
      W1, w1c, /*KT=*/18, /*NT=*/64, /*Kdim=*/KXd, /*Ndim=*/HHd);
  convert_weights_kernel<<<(PP * 32 * 32 * 256) / 256, 256, 0, stream>>>(
      W2, w2c, /*KT=*/32, /*NT=*/32, /*Kdim=*/HHd, /*Ndim=*/DD);

  // 3) fused grouped MLP (max tiles = 16384/64 + 7 pad tiles = 263)
  mlp_kernel<<<263, 512, LDSB, stream>>>(latents, actions, b1, b2,
                                         (const char*)w1c, (const char*)w2c,
                                         abase, perm, out);
}
